// lstm2_40656160424556
// MI455X (gfx1250) — matrix operand order repaired and checked by reading, not compiled, not run on hardware
//
#include <hip/hip_runtime.h>
#include <hip/hip_bf16.h>

// ---------------------------------------------------------------------------
// 2-layer LSTM (B=2048, T=128, D_IN=2, H1=128, H2=64) + FC(11), fully fused.
//
// One workgroup (8 waves / 256 thr) owns 16 batch rows for the whole sequence.
// All matmuls are v_wmma_f32_16x16x32_f16; weight B-fragments live in VGPRs
// for all 128 timesteps. Gate ownership is chosen so that i/f/g/o for a given
// hidden unit land in the SAME lane/element of the four WMMA accumulators:
//   wave w owns N-tiles { k0, k0+H, k0+2H, k0+3H }, k0 = w*16
// so activations + cell update run entirely in registers (no gate LDS tile).
// The x-projection (D_IN=2) is folded into the layer-1 WMMA via a K=160
// augmented contraction: h1 rows are [h(128) | x0 x1 | zeros], weights are
// [W_hh1 | W_ih1 | 0]. h1/h2 are double-buffered in LDS -> 2 barriers/step.
// Wave 4 streams x(t+1); wave 5 runs the FC WMMAs one step behind, off the
// critical path. The [B,8192] flat tensor never exists in memory.
// ---------------------------------------------------------------------------

typedef __attribute__((ext_vector_type(16))) _Float16 v16h;
typedef __attribute__((ext_vector_type(8)))  _Float16 v8h;
typedef __attribute__((ext_vector_type(8)))  float    v8f;

#define H1      128
#define H2      64
#define T_STEPS 128
#define NCLS    11

#define K1X     160            // augmented layer-1 contraction: 128 h + 2 x + 30 pad
#define HP1     (K1X + 8)      // 168 halves: row stride 336B (16B aligned, skewed)
#define HP2     (H2 + 8)       // 72 halves:  row stride 144B

// ws layout (_Float16 elements)
#define WS_WHH1X 0             // 512*160 = 81920  ([W_hh1 | W_ih1 | 0] rows)
#define WS_WIH2  81920         // 256*128 = 32768
#define WS_WHH2  114688        // 256*64  = 16384
#define WS_WFC   131072        // 128*16*64 = 131072 ([t][class(pad16)][k])
#define WS_TOTAL 262144        // 512 KB

__device__ __forceinline__ float sigm_f(float x) {
    return 1.0f / (1.0f + __expf(-x));
}
__device__ __forceinline__ float tanh_f(float x) {
    return 1.0f - 2.0f / (__expf(2.0f * x) + 1.0f);
}

// A-fragment (16x32 f16, M = lane&15), layout matching B:
//   elem j in 0..15 -> K = kbase + laneHi*16 + j
__device__ __forceinline__ v16h load_a_frag(const _Float16* row, int kbase, int lhi) {
    const _Float16* p = row + kbase + lhi * 16;
    v8h lo = *(const v8h*)(p);
    v8h hi = *(const v8h*)(p + 8);
    return __builtin_shufflevector(lo, hi, 0,1,2,3,4,5,6,7,8,9,10,11,12,13,14,15);
}

// B-fragment (32x16 f16, N = lane&15): elem j -> K = kbase + laneHi*16 + j;
// with W stored row-major [n][K] this is one contiguous 32B load per lane.
__device__ __forceinline__ v16h load_b_frag(const _Float16* Wrow_k0) {
    return *(const v16h*)Wrow_k0;
}

__device__ __forceinline__ v8f wmma_f16(v16h a, v16h b, v8f c) {
    return __builtin_amdgcn_wmma_f32_16x16x32_f16(false, a, false, b, (short)0, c,
                                                  false, false);
}

// ---------------------------------------------------------------------------
// Prep: build f16 weight images in ws.
// ---------------------------------------------------------------------------
__global__ void lstm2_prep_kernel(const float* __restrict__ Wih1,   // [512][2]
                                  const float* __restrict__ Whh1,   // [512][128]
                                  const float* __restrict__ Wih2,   // [256][128]
                                  const float* __restrict__ Whh2,   // [256][64]
                                  const float* __restrict__ Wfc,    // [11][8192]
                                  _Float16* __restrict__ ws) {
    for (int i = blockIdx.x * blockDim.x + threadIdx.x; i < WS_TOTAL;
         i += gridDim.x * blockDim.x) {
        if (i < WS_WIH2) {                        // augmented [W_hh1 | W_ih1 | 0]
            const int n = i / K1X, k = i % K1X;
            float v;
            if (k < H1)           v = Whh1[n * H1 + k];
            else if (k == H1)     v = Wih1[n * 2 + 0];
            else if (k == H1 + 1) v = Wih1[n * 2 + 1];
            else                  v = 0.0f;
            ws[i] = (_Float16)v;
        } else if (i < WS_WHH2) {
            ws[i] = (_Float16)Wih2[i - WS_WIH2];
        } else if (i < WS_WFC) {
            ws[i] = (_Float16)Whh2[i - WS_WHH2];
        } else {
            const int j = i - WS_WFC;
            const int t = j >> 10, n = (j >> 6) & 15, k = j & 63;
            ws[i] = (n < NCLS) ? (_Float16)Wfc[n * (T_STEPS * H2) + t * H2 + k]
                               : (_Float16)0.0f;
        }
    }
}

// ---------------------------------------------------------------------------
// Fused LSTM x2 + FC. Grid: B/16 blocks x 256 threads (8 waves).
// ---------------------------------------------------------------------------
__global__ void __launch_bounds__(256, 1)
lstm2_fused_kernel(const float* __restrict__ x,      // [B][2][T]
                   const float* __restrict__ bfc,    // [11]
                   const _Float16* __restrict__ ws,
                   float* __restrict__ out,          // [B][11]
                   int B) {
    __shared__ _Float16 h1s[2 * 16 * HP1];   // ping-pong [h | x0 x1 | 0] rows
    __shared__ _Float16 h2s[2 * 16 * HP2];

    const _Float16* whh1x = ws + WS_WHH1X;
    const _Float16* wih2h = ws + WS_WIH2;
    const _Float16* whh2h = ws + WS_WHH2;
    const _Float16* wfcp  = ws + WS_WFC;

    const int tid  = threadIdx.x;
    const int wave = tid >> 5;
    const int lane = tid & 31;
    const int l15  = lane & 15;
    const int lhi  = (lane >> 4) & 1;
    const int b0   = blockIdx.x * 16;
    const int w2   = wave & 3;

    // ---- init LDS state; seed x(t=0) into buffer 0
    for (int i = tid; i < 2 * 16 * HP1; i += 256) h1s[i] = (_Float16)0.0f;
    for (int i = tid; i < 2 * 16 * HP2; i += 256) h2s[i] = (_Float16)0.0f;
    __syncthreads();
    if (tid < 32) {
        const int m = tid >> 1, d = tid & 1;
        h1s[m * HP1 + H1 + d] = (_Float16)x[(b0 + m) * (2 * T_STEPS) + d * T_STEPS];
    }

    // ---- persistent weight B-fragments (VGPR-resident all 128 steps)
    // layer 1: wave w owns hidden k-range [w*16, w*16+16); gate tiles at
    //          n0 = gate*128 + w*16; K = 160 -> 5 k-steps.
    v16h b1[4][5];
#pragma unroll
    for (int g = 0; g < 4; ++g) {
        const int n = g * H1 + wave * 16 + l15;
#pragma unroll
        for (int ks = 0; ks < 5; ++ks)
            b1[g][ks] = load_b_frag(whh1x + n * K1X + ks * 32 + lhi * 16);
    }
    // layer 2 (waves 0..3 active): k2-range [w2*16, w2*16+16); gate tiles at
    //          n0 = gate*64 + w2*16; input K=128 (4 ks), recurrent K=64 (2 ks).
    v16h b2i[4][4], b2h[4][2];
#pragma unroll
    for (int g = 0; g < 4; ++g) {
        const int n = g * H2 + w2 * 16 + l15;
#pragma unroll
        for (int ks = 0; ks < 4; ++ks)
            b2i[g][ks] = load_b_frag(wih2h + n * H1 + ks * 32 + lhi * 16);
#pragma unroll
        for (int ks = 0; ks < 2; ++ks)
            b2h[g][ks] = load_b_frag(whh2h + n * H2 + ks * 32 + lhi * 16);
    }

    // per-lane cell state: element r <-> batch row m = lhi*8 + r
    float c1r[8] = {0.f,0.f,0.f,0.f,0.f,0.f,0.f,0.f};
    float c2r[8] = {0.f,0.f,0.f,0.f,0.f,0.f,0.f,0.f};
    v8f afc = {};                       // FC accumulator (wave 5)

    const int k1 = wave * 16 + l15;     // layer-1 hidden unit this lane produces
    const int k2 = w2 * 16 + l15;       // layer-2 hidden unit (waves 0..3)

    __syncthreads();

    for (int t = 0; t < T_STEPS; ++t) {
        const _Float16* h1cur = h1s + (t & 1) * (16 * HP1);
        _Float16*       h1nxt = h1s + ((t + 1) & 1) * (16 * HP1);
        const _Float16* h2cur = h2s + (t & 1) * (16 * HP2);
        _Float16*       h2nxt = h2s + ((t + 1) & 1) * (16 * HP2);

        // ============ phase A (all 8 waves): layer-1 WMMA + gates ============
        {
            v16h a[5];
#pragma unroll
            for (int ks = 0; ks < 5; ++ks)
                a[ks] = load_a_frag(h1cur + l15 * HP1, ks * 32, lhi);
            v8f acc[4] = {{}, {}, {}, {}};
#pragma unroll
            for (int g = 0; g < 4; ++g)
#pragma unroll
                for (int ks = 0; ks < 5; ++ks)
                    acc[g] = wmma_f16(a[ks], b1[g][ks], acc[g]);
            // gates fully in registers (i/f/g/o share lane & element index)
#pragma unroll
            for (int r = 0; r < 8; ++r) {
                const float iv = sigm_f(acc[0][r]);
                const float fv = sigm_f(acc[1][r]);
                const float gv = tanh_f(acc[2][r]);
                const float ov = sigm_f(acc[3][r]);
                c1r[r] = fv * c1r[r] + iv * gv;
                h1nxt[(lhi * 8 + r) * HP1 + k1] = (_Float16)(ov * tanh_f(c1r[r]));
            }
        }
        __syncthreads();

        // ===== phase B: layer-2 (waves 0-3) | x-stream (wave 4) | FC (wave 5) =
        if (wave < 4) {
            v16h ai[4], ah[2];
#pragma unroll
            for (int ks = 0; ks < 4; ++ks)
                ai[ks] = load_a_frag(h1nxt + l15 * HP1, ks * 32, lhi);  // h1_t
#pragma unroll
            for (int ks = 0; ks < 2; ++ks)
                ah[ks] = load_a_frag(h2cur + l15 * HP2, ks * 32, lhi);  // h2_{t-1}
            v8f acc[4] = {{}, {}, {}, {}};
#pragma unroll
            for (int g = 0; g < 4; ++g) {
#pragma unroll
                for (int ks = 0; ks < 4; ++ks)
                    acc[g] = wmma_f16(ai[ks], b2i[g][ks], acc[g]);
#pragma unroll
                for (int ks = 0; ks < 2; ++ks)
                    acc[g] = wmma_f16(ah[ks], b2h[g][ks], acc[g]);
            }
#pragma unroll
            for (int r = 0; r < 8; ++r) {
                const float iv = sigm_f(acc[0][r]);
                const float fv = sigm_f(acc[1][r]);
                const float gv = tanh_f(acc[2][r]);
                const float ov = sigm_f(acc[3][r]);
                c2r[r] = fv * c2r[r] + iv * gv;
                h2nxt[(lhi * 8 + r) * HP2 + k2] = (_Float16)(ov * tanh_f(c2r[r]));
            }
        } else if (wave == 4) {
            // stream x(t+1) into next h1 buffer's augmented columns
            const int m = lane >> 1, d = lane & 1;
            const int tp1 = (t + 1 < T_STEPS) ? (t + 1) : (T_STEPS - 1);
            h1nxt[m * HP1 + H1 + d] =
                (_Float16)x[(b0 + m) * (2 * T_STEPS) + d * T_STEPS + tp1];
        } else if (wave == 5) {
            // FC for step t-1 (h2_{t-1} = h2cur), off the critical path
            if (t > 0) {
#pragma unroll
                for (int ks = 0; ks < 2; ++ks) {
                    v16h af = load_a_frag(h2cur + l15 * HP2, ks * 32, lhi);
                    v16h bf = load_b_frag(wfcp + (t - 1) * (16 * H2) + l15 * H2 +
                                          ks * 32 + lhi * 16);
                    afc = wmma_f16(af, bf, afc);
                }
            }
        }
        __syncthreads();
    }

    // ---- tail: FC for t = 127 (h2_127 sits in buffer (T_STEPS & 1) == 0)
    if (wave == 5) {
        const _Float16* h2fc = h2s + (T_STEPS & 1) * (16 * HP2);
#pragma unroll
        for (int ks = 0; ks < 2; ++ks) {
            v16h af = load_a_frag(h2fc + l15 * HP2, ks * 32, lhi);
            v16h bf = load_b_frag(wfcp + (T_STEPS - 1) * (16 * H2) + l15 * H2 +
                                  ks * 32 + lhi * 16);
            afc = wmma_f16(af, bf, afc);
        }
        if (l15 < NCLS) {
#pragma unroll
            for (int r = 0; r < 8; ++r) {
                const int m = lhi * 8 + r;
                out[(b0 + m) * NCLS + l15] = afc[r] + bfc[l15];
            }
        }
    }
    (void)B;
}

extern "C" void kernel_launch(void* const* d_in, const int* in_sizes, int n_in,
                              void* d_out, int out_size, void* d_ws, size_t ws_size,
                              hipStream_t stream) {
    const float* x    = (const float*)d_in[0];   // [B,2,128]
    const float* Wih1 = (const float*)d_in[1];   // [512,2]
    const float* Whh1 = (const float*)d_in[2];   // [512,128]
    const float* Wih2 = (const float*)d_in[3];   // [256,128]
    const float* Whh2 = (const float*)d_in[4];   // [256,64]
    const float* Wfc  = (const float*)d_in[5];   // [11,8192]
    const float* bfc  = (const float*)d_in[6];   // [11]
    float* out = (float*)d_out;
    _Float16* ws = (_Float16*)d_ws;

    const int B = in_sizes[0] / (2 * T_STEPS);   // 2048

    lstm2_prep_kernel<<<(WS_TOTAL + 255) / 256, 256, 0, stream>>>(Wih1, Whh1, Wih2,
                                                                  Whh2, Wfc, ws);
    lstm2_fused_kernel<<<B / 16, 256, 0, stream>>>(x, bfc, ws, out, B);
    (void)n_in; (void)out_size; (void)ws_size;
}
